// Zoomer_68504728371218
// MI455X (gfx1250) — matrix operand order repaired and checked by reading, not compiled, not run on hardware
//
#include <hip/hip_runtime.h>
#include <hip/hip_bf16.h>

typedef _Float16 h16;
typedef __attribute__((ext_vector_type(4)))  _Float16 v4h;
typedef __attribute__((ext_vector_type(16))) _Float16 v16h;
typedef __attribute__((ext_vector_type(8)))  float    v8f;

// ---------------------------------------------------------------- WMMA core
__device__ __forceinline__ v8f wmma32(v16h a, v16h b, v8f c) {
  return __builtin_amdgcn_wmma_f32_16x16x32_f16(false, a, false, b, (short)0, c,
                                                false, false);
}

// single-instruction ReLU: v_med3_f32(v, 0, +FLT_MAX)
__device__ __forceinline__ float relu1(float v) {
  return __builtin_amdgcn_fmed3f(v, 0.f, 3.402823466e38f);
}

// Bias folded into the C operand: every accumulator element of a lane maps to
// column nt+(lane&15), so C = splat(bias[col]) makes D = X*W + b exactly.
__device__ __forceinline__ v8f bias8(const float* bias, int nt, int lane) {
  float bv = bias[nt + (lane & 15)];
  v8f a = {bv, bv, bv, bv, bv, bv, bv, bv};
  return a;
}

// A fragment: X row-major in LDS (ldx halves/row), rows 0..15, K-chunk at kof.
// lane<16 -> row=lane, K=kof+{0..7,16..23}; lane>=16 -> row=lane-16, K=kof+{8..15,24..31}
__device__ __forceinline__ v16h load_A16(const h16* X, int ldx, int kof, int lane) {
  int r  = lane & 15;
  int kb = kof + ((lane < 16) ? 0 : 8);
  const h16* p = X + r * ldx + kb;
  v16h a;
#pragma unroll
  for (int i = 0; i < 8; ++i) { a[i] = p[i]; a[8 + i] = p[16 + i]; }
  return a;
}

// Stage W (K x N, f32, row-major) into per-lane fragment-ready f16 layout:
// frag f = kc*ntl + t ; lane l elem i = W[(kc*32 + ((l<16)?0:8) + (i&7) + ((i&8)?16:0)) * N + t*16 + (l&15)]
__device__ void stage_Bfrag(h16* WF, const float* W, int N, int nkc, int ntl) {
  int total = nkc * ntl * 512;
  for (int idx = threadIdx.x; idx < total; idx += blockDim.x) {
    int i = idx & 15, l = (idx >> 4) & 31, f = idx >> 9;
    int t = f % ntl, kc = f / ntl;
    int n = t * 16 + (l & 15);
    int k = kc * 32 + ((l < 16) ? 0 : 8) + (i & 7) + ((i & 8) ? 16 : 0);
    WF[idx] = (h16)W[k * N + n];
  }
}
// B fragment load: 16 contiguous halves (2 x ds_load_b128)
__device__ __forceinline__ v16h load_Bf(const h16* WF, int f, int lane) {
  const h16* p = WF + ((f << 5) + lane) * 16;
  v16h b;
#pragma unroll
  for (int i = 0; i < 16; ++i) b[i] = p[i];
  return b;
}

// D layout: lane l, elem j -> row (j + (l<16?0:8)), col nt+(l&15)
__device__ __forceinline__ void store_tile_lds_h(h16* Y, int ldy, int nt, v8f d,
                                                 bool do_relu, int lane) {
  int c  = nt + (lane & 15);
  int rb = (lane < 16) ? 0 : 8;
#pragma unroll
  for (int j = 0; j < 8; ++j) {
    float v = d[j];
    if (do_relu) v = relu1(v);
    Y[(rb + j) * ldy + c] = (h16)v;
  }
}

__device__ __forceinline__ void store_tile_glb(float* Y, int row0, int ldy, int nt,
                                               v8f d, bool do_relu, int lane) {
  int c  = nt + (lane & 15);
  int rb = (lane < 16) ? 0 : 8;
#pragma unroll
  for (int j = 0; j < 8; ++j) {
    float v = d[j];
    if (do_relu) v = relu1(v);
    Y[(unsigned)((row0 + rb + j) * ldy + c)] = v;
  }
}

__device__ __forceinline__ void lds_f32(float* dst, const float* src, int n) {
  for (int i = threadIdx.x; i < n; i += blockDim.x) dst[i] = src[i];
}
__device__ __forceinline__ v4h cvt4(float4 v) {
  v4h h; h[0] = (h16)v.x; h[1] = (h16)v.y; h[2] = (h16)v.z; h[3] = (h16)v.w;
  return h;
}
__device__ __forceinline__ float dot4(v4h a, float4 b) {
  return (float)a[0] * b.x + (float)a[1] * b.y + (float)a[2] * b.z + (float)a[3] * b.w;
}
__device__ __forceinline__ float dot4h(v4h a, v4h b) {
  return (float)a[0] * (float)b[0] + (float)a[1] * (float)b[1] +
         (float)a[2] * (float)b[2] + (float)a[3] * (float)b[3];
}

// --------------------------------------------------- gather / embedding-bag
__global__ void k_gather(const float* __restrict__ tab, const int* __restrict__ ids,
                         float* __restrict__ Y, int R) {
  unsigned i = blockIdx.x * blockDim.x + threadIdx.x;
  if (i >= (unsigned)(R * 16)) return;
  unsigned r = i >> 4, c4 = i & 15;
  ((float4*)Y)[i] = ((const float4*)tab)[(unsigned)ids[r] * 16u + c4];
}

__global__ void k_bag(const int* __restrict__ gids, const float* __restrict__ gtab,
                      float* __restrict__ Y, int R) {
  unsigned i = blockIdx.x * blockDim.x + threadIdx.x;
  if (i >= (unsigned)(R * 16)) return;
  unsigned r = i >> 4, c4 = i & 15;
  const int* g = gids + r * 3;
  float sx = 0.f, sy = 0.f, sz = 0.f, sw = 0.f, cnt = 0.f;
#pragma unroll
  for (int j = 0; j < 3; ++j) {
    unsigned id = (unsigned)g[j];
    if (id != 0) {
      float4 e = ((const float4*)gtab)[id * 16u + c4];
      sx += e.x; sy += e.y; sz += e.z; sw += e.w; cnt += 1.f;
    }
  }
  float inv = 1.f / fmaxf(cnt, 1.f);
  ((float4*)Y)[i] = make_float4(sx * inv, sy * inv, sz * inv, sw * inv);
}

// --------------------------------------------- gather + MLP(64 ->relu 32 -> 64)
__global__ __launch_bounds__(256) void k_mlp64(
    const float* __restrict__ tab, const int* __restrict__ ids, int id_stride,
    const float* __restrict__ W1, const float* __restrict__ b1,
    const float* __restrict__ W2, const float* __restrict__ b2,
    float* __restrict__ Y, int R) {
  __shared__ __align__(16) h16 W1F[4 * 512], W2F[4 * 512];
  __shared__ float b1s[32], b2s[64];
  __shared__ __align__(16) h16 Xs[8][16 * 64], Hs[8][16 * 32];
  stage_Bfrag(W1F, W1, 32, 2, 2);
  stage_Bfrag(W2F, W2, 64, 1, 4);
  lds_f32(b1s, b1, 32); lds_f32(b2s, b2, 64);
  __syncthreads();
  int lane = threadIdx.x & 31, wv = threadIdx.x >> 5;
  v16h w1f[4], w2f[4];
#pragma unroll
  for (int f = 0; f < 4; ++f) { w1f[f] = load_Bf(W1F, f, lane); w2f[f] = load_Bf(W2F, f, lane); }
  const float4* tab4 = (const float4*)tab;
  int ntile = R >> 4;
  for (int tile = blockIdx.x * 8 + wv; tile < ntile; tile += gridDim.x * 8) {
    int rbase = tile << 4;
    h16* X = Xs[wv]; h16* H = Hs[wv];
    for (int idx = lane; idx < 16 * 16; idx += 32) {
      int r = idx >> 4, c4 = idx & 15;
      unsigned id = (unsigned)ids[(rbase + r) * id_stride];
      *(v4h*)(X + r * 64 + c4 * 4) = cvt4(tab4[id * 16u + c4]);
    }
    v16h a0 = load_A16(X, 64, 0, lane), a1 = load_A16(X, 64, 32, lane);
#pragma unroll
    for (int t = 0; t < 2; ++t) {
      v8f acc = bias8(b1s, t * 16, lane);
      acc = wmma32(a0, w1f[t], acc);
      acc = wmma32(a1, w1f[2 + t], acc);
      store_tile_lds_h(H, 32, t * 16, acc, true, lane);
    }
    v16h ah = load_A16(H, 32, 0, lane);
#pragma unroll
    for (int t = 0; t < 4; ++t) {
      v8f acc = bias8(b2s, t * 16, lane);
      acc = wmma32(ah, w2f[t], acc);
      store_tile_glb(Y, rbase, 64, t * 16, acc, false, lane);
    }
  }
}

// ------------------------------------- Y = relu((A [+ Badd]) * W(64x64) + b)
__global__ __launch_bounds__(256) void k_linear64(
    const float* __restrict__ A, const float* __restrict__ Badd,
    const float* __restrict__ W, const float* __restrict__ b,
    float* __restrict__ Y, int R) {
  __shared__ __align__(16) h16 WF[8 * 512];
  __shared__ float bs[64];
  __shared__ __align__(16) h16 Xs[8][16 * 64];
  stage_Bfrag(WF, W, 64, 2, 4);
  lds_f32(bs, b, 64);
  __syncthreads();
  int lane = threadIdx.x & 31, wv = threadIdx.x >> 5;
  v16h wf[8];
#pragma unroll
  for (int f = 0; f < 8; ++f) wf[f] = load_Bf(WF, f, lane);
  const float4* A4 = (const float4*)A;
  const float4* B4 = (const float4*)Badd;
  int ntile = R >> 4;
  for (int tile = blockIdx.x * 8 + wv; tile < ntile; tile += gridDim.x * 8) {
    int rbase = tile << 4;
    h16* X = Xs[wv];
    for (int idx = lane; idx < 16 * 16; idx += 32) {
      int r = idx >> 4, c4 = idx & 15;
      unsigned o = (unsigned)((rbase + r) * 16 + c4);
      float4 v = A4[o];
      if (Badd) {
        float4 w2 = B4[o];
        v.x += w2.x; v.y += w2.y; v.z += w2.z; v.w += w2.w;
      }
      *(v4h*)(X + r * 64 + c4 * 4) = cvt4(v);
    }
    v16h a0 = load_A16(X, 64, 0, lane), a1 = load_A16(X, 64, 32, lane);
#pragma unroll
    for (int t = 0; t < 4; ++t) {
      v8f acc = bias8(bs, t * 16, lane);
      acc = wmma32(a0, wf[t], acc);
      acc = wmma32(a1, wf[4 + t], acc);
      store_tile_glb(Y, rbase, 64, t * 16, acc, true, lane);
    }
  }
}

// ------------- attention core, L=2 kv tokens per pair, 16 pairs per wave.
__global__ __launch_bounds__(64) void k_attn2(
    const float* __restrict__ qp, int qp_shift,
    const int* __restrict__ ids, int id_stride, const float* __restrict__ tab,
    const float* __restrict__ row0, const float* __restrict__ row1,
    const float* __restrict__ Wk, const float* __restrict__ bk,
    const float* __restrict__ Wv, const float* __restrict__ bv,
    const float* __restrict__ Wo, const float* __restrict__ bo,
    float* __restrict__ Y, int npairs) {
  __shared__ __align__(16) h16 WkF[8 * 512], WvF[8 * 512], WoF[8 * 512];
  __shared__ float bks[64], bvs[64], bos[64];
  __shared__ __align__(16) h16 Xs[2][32 * 64], Ks[2][32 * 64];
  __shared__ __align__(16) h16 QPs[2][16 * 64], CTs[2][16 * 64];
  __shared__ float aSs[2][32];
  stage_Bfrag(WkF, Wk, 64, 2, 4);
  stage_Bfrag(WvF, Wv, 64, 2, 4);
  stage_Bfrag(WoF, Wo, 64, 2, 4);
  lds_f32(bks, bk, 64); lds_f32(bvs, bv, 64); lds_f32(bos, bo, 64);
  __syncthreads();
  int lane = threadIdx.x & 31, wv = threadIdx.x >> 5;
  const float4* tab4 = (const float4*)tab;
  const float4* r04  = (const float4*)row0;
  const float4* r14  = (const float4*)row1;
  const float4* qp4  = (const float4*)qp;
  int ntile = npairs >> 4;
  for (int tile = blockIdx.x * 2 + wv; tile < ntile; tile += gridDim.x * 2) {
    int pbase = tile << 4;
    h16* X = Xs[wv]; h16* K = Ks[wv]; h16* QP = QPs[wv]; h16* CT = CTs[wv];
    float* aS = aSs[wv];
    for (int idx = lane; idx < 32 * 16; idx += 32) {      // gather 32 kv rows
      int r = idx >> 4, c4 = idx & 15;
      unsigned p = (unsigned)(pbase + (r >> 1));
      float4 v;
      if ((r & 1) == 0)
        v = ids ? tab4[(unsigned)ids[p * id_stride] * 16u + c4] : r04[p * 16u + c4];
      else
        v = r14[p * 16u + c4];
      *(v4h*)(X + r * 64 + c4 * 4) = cvt4(v);
    }
    for (int idx = lane; idx < 16 * 16; idx += 32) {      // per-pair q' rows
      int r = idx >> 4, c4 = idx & 15;
      unsigned p = (unsigned)(pbase + r);
      *(v4h*)(QP + r * 64 + c4 * 4) = cvt4(qp4[(p >> qp_shift) * 16u + c4]);
    }
    // A fragments of X, reused by both the K and V passes
    v16h af[4];
    af[0] = load_A16(X, 64, 0, lane);        af[1] = load_A16(X, 64, 32, lane);
    af[2] = load_A16(X + 1024, 64, 0, lane); af[3] = load_A16(X + 1024, 64, 32, lane);
    // K' = relu(X*Wk + bk) -> LDS
#pragma unroll
    for (int rt = 0; rt < 2; ++rt) {
#pragma unroll
      for (int t = 0; t < 4; ++t) {
        v8f acc = bias8(bks, t * 16, lane);
        acc = wmma32(af[rt * 2], load_Bf(WkF, t, lane), acc);
        acc = wmma32(af[rt * 2 + 1], load_Bf(WkF, 4 + t, lane), acc);
        store_tile_lds_h(K + rt * 16 * 64, 64, t * 16, acc, true, lane);
      }
    }
    // scores + 2-way softmax (lane <-> kv row)
    float s = 0.f;
    {
      const v4h* kr = (const v4h*)(K + lane * 64);
      const v4h* qr = (const v4h*)(QP + (lane >> 1) * 64);
#pragma unroll
      for (int c4 = 0; c4 < 16; ++c4) s += dot4h(kr[c4], qr[c4]);
      s *= 0.125f;
    }
    float so = __shfl_xor(s, 1, 32);
    float m  = fmaxf(s, so);
    float e  = __expf(s - m), eo = __expf(so - m);
    aS[lane] = e / (e + eo);
    // V' pass fused with att@V: both rows of a pair live in one lane's acc
#pragma unroll
    for (int rt = 0; rt < 2; ++rt) {
      int rb = rt * 16 + ((lane < 16) ? 0 : 8);
#pragma unroll
      for (int t = 0; t < 4; ++t) {
        v8f acc = bias8(bvs, t * 16, lane);
        acc = wmma32(af[rt * 2], load_Bf(WvF, t, lane), acc);
        acc = wmma32(af[rt * 2 + 1], load_Bf(WvF, 4 + t, lane), acc);
        int c = t * 16 + (lane & 15);
#pragma unroll
        for (int j = 0; j < 8; j += 2) {
          int row = rb + j;
          CT[(row >> 1) * 64 + c] =
              (h16)(aS[row] * relu1(acc[j]) + aS[row + 1] * relu1(acc[j + 1]));
        }
      }
    }
    // O = relu(CT*Wo + bo)
    v16h c0 = load_A16(CT, 64, 0, lane), c1 = load_A16(CT, 64, 32, lane);
#pragma unroll
    for (int t = 0; t < 4; ++t) {
      v8f acc = bias8(bos, t * 16, lane);
      acc = wmma32(c0, load_Bf(WoF, t, lane), acc);
      acc = wmma32(c1, load_Bf(WoF, 4 + t, lane), acc);
      store_tile_glb(Y, pbase, 64, t * 16, acc, true, lane);
    }
  }
}

// --------------- GAT attention: kv = [nbr(b,32,64); self(b,64)], one b per wave
__global__ __launch_bounds__(64) void k_attn33(
    const float* __restrict__ qp, const float* __restrict__ nbr,
    const float* __restrict__ selfr,
    const float* __restrict__ Wk, const float* __restrict__ bk,
    const float* __restrict__ Wv, const float* __restrict__ bv,
    const float* __restrict__ Wo, const float* __restrict__ bo,
    float* __restrict__ Y, int Bn) {
  __shared__ __align__(16) h16 WkF[8 * 512], WvF[8 * 512], WoT[4096];
  __shared__ float bks[64], bvs[64], bos[64];
  __shared__ __align__(16) h16 Xs[2][48 * 64], Ks[2][48 * 64];
  __shared__ __align__(16) float qs[2][64], cxs[2][64];
  __shared__ float sS[2][48], aS[2][48];
  stage_Bfrag(WkF, Wk, 64, 2, 4);
  stage_Bfrag(WvF, Wv, 64, 2, 4);
  for (int idx = threadIdx.x; idx < 4096; idx += blockDim.x) {  // Wo transposed
    int c = idx >> 6, k = idx & 63;
    WoT[idx] = (h16)Wo[k * 64 + c];
  }
  lds_f32(bks, bk, 64); lds_f32(bvs, bv, 64); lds_f32(bos, bo, 64);
  __syncthreads();
  int lane = threadIdx.x & 31, wv = threadIdx.x >> 5;
  const float4* nbr4  = (const float4*)nbr;
  const float4* self4 = (const float4*)selfr;
  for (int b = blockIdx.x * 2 + wv; b < Bn; b += gridDim.x * 2) {
    h16* X = Xs[wv]; h16* K = Ks[wv];
    float* q = qs[wv]; float* sL = sS[wv]; float* aL = aS[wv]; float* cx = cxs[wv];
    for (int c4 = lane; c4 < 16; c4 += 32)
      ((float4*)q)[c4] = ((const float4*)qp)[(unsigned)b * 16u + c4];
    for (int idx = lane; idx < 48 * 16; idx += 32) {
      int r = idx >> 4, c4 = idx & 15;
      float4 v = make_float4(0.f, 0.f, 0.f, 0.f);
      if (r < 32)       v = nbr4[(unsigned)(b * 32 + r) * 16u + c4];
      else if (r == 32) v = self4[(unsigned)b * 16u + c4];
      *(v4h*)(X + r * 64 + c4 * 4) = cvt4(v);
    }
    // A fragments of X (3 row tiles), reused by K and V passes
    v16h af[6];
#pragma unroll
    for (int rt = 0; rt < 3; ++rt) {
      af[rt * 2]     = load_A16(X + rt * 16 * 64, 64, 0, lane);
      af[rt * 2 + 1] = load_A16(X + rt * 16 * 64, 64, 32, lane);
    }
    // K' = relu(X*Wk + bk)
#pragma unroll
    for (int rt = 0; rt < 3; ++rt) {
#pragma unroll
      for (int t = 0; t < 4; ++t) {
        v8f acc = bias8(bks, t * 16, lane);
        acc = wmma32(af[rt * 2], load_Bf(WkF, t, lane), acc);
        acc = wmma32(af[rt * 2 + 1], load_Bf(WkF, 4 + t, lane), acc);
        store_tile_lds_h(K + rt * 16 * 64, 64, t * 16, acc, true, lane);
      }
    }
    // scores over 33 valid rows; masked softmax
    for (int j = lane; j < 48; j += 32) aL[j] = 0.f;
    for (int j = lane; j < 33; j += 32) {
      float s = 0.f;
      const v4h* kr = (const v4h*)(K + j * 64);
      const float4* q4 = (const float4*)q;
#pragma unroll
      for (int c4 = 0; c4 < 16; ++c4) s += dot4(kr[c4], q4[c4]);
      sL[j] = s * 0.125f;
    }
    float m = -1e30f;
    for (int j = 0; j <= 32; ++j) m = fmaxf(m, sL[j]);
    float sum = 0.f;
    for (int j = 0; j <= 32; ++j) sum += __expf(sL[j] - m);
    for (int j = lane; j < 33; j += 32) aL[j] = __expf(sL[j] - m) / sum;
    // V' pass fused with att@V reduction: ctx[c] = sum_j aL[j]*relu(V[j][c]+bv)
#pragma unroll
    for (int t = 0; t < 4; ++t) {
      int c = t * 16 + (lane & 15);
      float pc = 0.f;
#pragma unroll
      for (int rt = 0; rt < 3; ++rt) {
        v8f acc = bias8(bvs, t * 16, lane);
        acc = wmma32(af[rt * 2], load_Bf(WvF, t, lane), acc);
        acc = wmma32(af[rt * 2 + 1], load_Bf(WvF, 4 + t, lane), acc);
        int rb = rt * 16 + ((lane < 16) ? 0 : 8);
#pragma unroll
        for (int j = 0; j < 8; ++j)
          pc += aL[rb + j] * relu1(acc[j]);
      }
      pc += __shfl_xor(pc, 16, 32);
      if (lane < 16) cx[c] = pc;
    }
    // O = relu(ctx * Wo + bo) (1x64, Wo transposed for contiguous reads)
    for (int c = lane; c < 64; c += 32) {
      float o = bos[c];
      const v4h* wr = (const v4h*)(WoT + c * 64);
      const float4* cx4 = (const float4*)cx;
#pragma unroll
      for (int k4 = 0; k4 < 16; ++k4) o += dot4(wr[k4], cx4[k4]);
      Y[(unsigned)(b * 64 + c)] = relu1(o);
    }
  }
}

// ----------------- MLP on concat([A,Bc],-1): 128 ->relu 64 -> 64 (no out relu)
__global__ __launch_bounds__(128) void k_mlp_cat(
    const float* __restrict__ A, const float* __restrict__ Bc,
    const float* __restrict__ W1, const float* __restrict__ b1,
    const float* __restrict__ W2, const float* __restrict__ b2,
    float* __restrict__ Y, int R) {
  __shared__ __align__(16) h16 W1F[16 * 512], W2F[8 * 512];
  __shared__ float b1s[64], b2s[64];
  __shared__ __align__(16) h16 Xs[4][16 * 128], Hs[4][16 * 64];
  stage_Bfrag(W1F, W1, 64, 4, 4);
  stage_Bfrag(W2F, W2, 64, 2, 4);
  lds_f32(b1s, b1, 64); lds_f32(b2s, b2, 64);
  __syncthreads();
  int lane = threadIdx.x & 31, wv = threadIdx.x >> 5;
  const float4* A4 = (const float4*)A;
  const float4* B4 = (const float4*)Bc;
  int ntile = R >> 4;
  for (int tile = blockIdx.x * 4 + wv; tile < ntile; tile += gridDim.x * 4) {
    int rbase = tile << 4;
    h16* X = Xs[wv]; h16* H = Hs[wv];
    for (int idx = lane; idx < 16 * 32; idx += 32) {
      int r = idx >> 5, c4 = idx & 31;
      float4 v = (c4 < 16) ? A4[(unsigned)(rbase + r) * 16u + c4]
                           : B4[(unsigned)(rbase + r) * 16u + (c4 - 16)];
      *(v4h*)(X + r * 128 + c4 * 4) = cvt4(v);
    }
    v16h af[4];
#pragma unroll
    for (int kc = 0; kc < 4; ++kc) af[kc] = load_A16(X, 128, kc * 32, lane);
#pragma unroll
    for (int t = 0; t < 4; ++t) {
      v8f acc = bias8(b1s, t * 16, lane);
#pragma unroll
      for (int kc = 0; kc < 4; ++kc)
        acc = wmma32(af[kc], load_Bf(W1F, kc * 4 + t, lane), acc);
      store_tile_lds_h(H, 64, t * 16, acc, true, lane);
    }
    v16h h0 = load_A16(H, 64, 0, lane), h1 = load_A16(H, 64, 32, lane);
#pragma unroll
    for (int t = 0; t < 4; ++t) {
      v8f acc = bias8(b2s, t * 16, lane);
      acc = wmma32(h0, load_Bf(W2F, t, lane), acc);
      acc = wmma32(h1, load_Bf(W2F, 4 + t, lane), acc);
      store_tile_glb(Y, rbase, 64, t * 16, acc, false, lane);
    }
  }
}

// -------- final DSSM head: x=[qu, ie, qu*ie](192) ->relu 96 -> 1 -> sigmoid
__global__ __launch_bounds__(64) void k_dssm(
    const float* __restrict__ QU, const float* __restrict__ IE,
    const float* __restrict__ W1, const float* __restrict__ b1,
    const float* __restrict__ W2, const float* __restrict__ b2,
    float* __restrict__ out, int R) {
  __shared__ __align__(16) h16 W1F[36 * 512];
  __shared__ __align__(16) float W2s[96];
  __shared__ float b1s[96], b2s;
  __shared__ __align__(16) h16 Xs[2][16 * 192], Hs[2][16 * 96];
  stage_Bfrag(W1F, W1, 96, 6, 6);
  lds_f32(W2s, W2, 96); lds_f32(b1s, b1, 96);
  if (threadIdx.x == 0) b2s = b2[0];
  __syncthreads();
  int lane = threadIdx.x & 31, wv = threadIdx.x >> 5;
  const float4* QU4 = (const float4*)QU;
  const float4* IE4 = (const float4*)IE;
  int ntile = R >> 4;
  for (int tile = blockIdx.x * 2 + wv; tile < ntile; tile += gridDim.x * 2) {
    int rbase = tile << 4;
    h16* X = Xs[wv]; h16* H = Hs[wv];
    for (int idx = lane; idx < 16 * 48; idx += 32) {
      int r = idx / 48, c4 = idx - r * 48;
      unsigned ro = (unsigned)(rbase + r) * 16u;
      float4 v;
      if (c4 < 16) v = QU4[ro + c4];
      else if (c4 < 32) v = IE4[ro + (c4 - 16)];
      else {
        float4 a = QU4[ro + (c4 - 32)];
        float4 bb = IE4[ro + (c4 - 32)];
        v = make_float4(a.x * bb.x, a.y * bb.y, a.z * bb.z, a.w * bb.w);
      }
      *(v4h*)(X + r * 192 + c4 * 4) = cvt4(v);
    }
    v16h af[6];
#pragma unroll
    for (int kc = 0; kc < 6; ++kc) af[kc] = load_A16(X, 192, kc * 32, lane);
#pragma unroll
    for (int t = 0; t < 6; ++t) {
      v8f acc = bias8(b1s, t * 16, lane);
#pragma unroll
      for (int kc = 0; kc < 6; ++kc)
        acc = wmma32(af[kc], load_Bf(W1F, kc * 6 + t, lane), acc);
      store_tile_lds_h(H, 96, t * 16, acc, true, lane);
    }
    if (lane < 16) {
      float v = b2s;
      const v4h* hr = (const v4h*)(H + lane * 96);
      const float4* w4 = (const float4*)W2s;
#pragma unroll
      for (int k4 = 0; k4 < 24; ++k4) v += dot4(hr[k4], w4[k4]);
      out[(unsigned)(rbase + lane)] = 1.f / (1.f + __expf(-v));
    }
  }
}

// =========================================================================
extern "C" void kernel_launch(void* const* d_in, const int* in_sizes, int n_in,
                              void* d_out, int out_size, void* d_ws, size_t ws_size,
                              hipStream_t stream) {
  (void)in_sizes; (void)n_in; (void)out_size; (void)ws_size;
  const int B = 8192, NBn = 32;
  const int* uids   = (const int*)d_in[0];
  const int* mids   = (const int*)d_in[1];
  const int* qids   = (const int*)d_in[2];
  const int* mgenre = (const int*)d_in[3];
  const int* m_qids = (const int*)d_in[5];   // (B,NB,2), use [:,:,0]
  const int* m_uids = (const int*)d_in[6];
  const int* u_mids = (const int*)d_in[7];
  const int* q_mids = (const int*)d_in[8];
  const int* u_mgen = (const int*)d_in[9];
  const int* q_mgen = (const int*)d_in[11];
  // params: jax pytree leaves, dict keys sorted at every level, start at 13
#define P(i) ((const float*)d_in[i])
  const float *dssm_W1 = P(13), *dssm_W2 = P(14), *dssm_b1 = P(15), *dssm_b2 = P(16);
  const float *fla_Wk = P(17), *fla_Wo = P(18), *fla_Wq = P(19), *fla_Wv = P(20);
  const float *fla_bk = P(21), *fla_bo = P(22), *fla_bq = P(23), *fla_bv = P(24);
  const float *fmq_W1 = P(25), *fmq_W2 = P(26), *fmq_b1 = P(27), *fmq_b2 = P(28);
  const float *fmu_W1 = P(29), *fmu_W2 = P(30), *fmu_b1 = P(31), *fmu_b2 = P(32);
  const float *gat_Wk = P(33), *gat_Wo = P(34), *gat_Wq = P(35), *gat_Wv = P(36);
  const float *gat_bk = P(37), *gat_bo = P(38), *gat_bq = P(39), *gat_bv = P(40);
  const float *genre_tab = P(41), *item_tab = P(42);
  const float *out_W1 = P(43), *out_W2 = P(44), *out_b1 = P(45), *out_b2 = P(46);
  const float *qu_W1 = P(47), *qu_W2 = P(48), *qu_b1 = P(49), *qu_b2 = P(50);
  const float *query_tab = P(51);
  const float *sem_Wk = P(52), *sem_Wo = P(53), *sem_Wq = P(54), *sem_Wv = P(55);
  const float *sem_bk = P(56), *sem_bo = P(57), *sem_bq = P(58), *sem_bv = P(59);
  const float *user_tab = P(60);
#undef P
  // workspace layout (floats); total ~420 MB
  float* ws = (float*)d_ws;
  const long long BE = (long long)B * 64, BNE = (long long)B * NBn * 64;
  float *UF = ws,           *QF = ws + BE,      *IIF = ws + 2 * BE, *MGF = ws + 3 * BE;
  float *QPq = ws + 4 * BE, *QPu = ws + 5 * BE, *QPg = ws + 6 * BE, *IFt = ws + 7 * BE;
  float *QPig = ws + 8 * BE, *QPsm = ws + 9 * BE, *AU = ws + 10 * BE, *AQ = ws + 11 * BE;
  float *IE = ws + 12 * BE, *AGu = ws + 13 * BE, *AGq = ws + 14 * BE;
  float *UE = ws + 15 * BE, *QE = ws + 16 * BE,  *QU = ws + 17 * BE;
  float *big = ws + 18 * BE;
  float *GFu = big, *GFq = big + BNE, *MQF = big + 2 * BNE, *MUF = big + 3 * BNE;
  float *QIF = big + 4 * BNE, *UIF = big + 5 * BNE;

  const int RN = B * NBn;
  int gB  = (B * 16 + 255) / 256, gBN = (RN * 16 + 255) / 256;
  int t8B = ((B + 15) / 16 + 7) / 8, t8N = ((RN + 15) / 16 + 7) / 8;
  int t2B = ((B + 15) / 16 + 1) / 2, t2N = ((RN + 15) / 16 + 1) / 2;
  int t4B = ((B + 15) / 16 + 3) / 4, t33 = (B + 1) / 2;

  // embeddings / bags
  k_gather<<<gB, 256, 0, stream>>>(item_tab, mids, IIF, B);
  k_bag<<<gB, 256, 0, stream>>>(mgenre, genre_tab, MGF, B);
  k_bag<<<gBN, 256, 0, stream>>>(u_mgen, genre_tab, GFu, RN);
  k_bag<<<gBN, 256, 0, stream>>>(q_mgen, genre_tab, GFq, RN);
  // feature MLPs
  k_mlp64<<<t8B, 256, 0, stream>>>(user_tab, uids, 1, fmu_W1, fmu_b1, fmu_W2, fmu_b2, UF, B);
  k_mlp64<<<t8B, 256, 0, stream>>>(query_tab, qids, 1, fmq_W1, fmq_b1, fmq_W2, fmq_b2, QF, B);
  k_mlp64<<<t8N, 256, 0, stream>>>(query_tab, m_qids, 2, fmq_W1, fmq_b1, fmq_W2, fmq_b2, MQF, RN);
  k_mlp64<<<t8N, 256, 0, stream>>>(user_tab, m_uids, 2, fmu_W1, fmu_b1, fmu_W2, fmu_b2, MUF, RN);
  // q projections (fla + gat; (node+roi) symmetric -> one shared gat proj)
  k_linear64<<<t8B, 256, 0, stream>>>(QF, nullptr, fla_Wq, fla_bq, QPq, B);
  k_linear64<<<t8B, 256, 0, stream>>>(UF, nullptr, fla_Wq, fla_bq, QPu, B);
  k_linear64<<<t8B, 256, 0, stream>>>(UF, QF, gat_Wq, gat_bq, QPg, B);
  // feature-level attention (L=2); qp_shift: 5 -> qp row = pair/32, 0 -> per pair
  k_attn2<<<t2N, 64, 0, stream>>>(QPq, 5, q_mids, 2, item_tab, nullptr, GFq,
                                  fla_Wk, fla_bk, fla_Wv, fla_bv, fla_Wo, fla_bo, QIF, RN);
  k_attn2<<<t2N, 64, 0, stream>>>(QPu, 5, u_mids, 2, item_tab, nullptr, GFu,
                                  fla_Wk, fla_bk, fla_Wv, fla_bv, fla_Wo, fla_bo, UIF, RN);
  k_attn2<<<t2B, 64, 0, stream>>>(QPu, 0, nullptr, 0, nullptr, IIF, MGF,
                                  fla_Wk, fla_bk, fla_Wv, fla_bv, fla_Wo, fla_bo, IFt, B);
  // user / query graph aggregate (GAT over 33)
  k_attn33<<<t33, 64, 0, stream>>>(QPg, UIF, UF, gat_Wk, gat_bk, gat_Wv, gat_bv,
                                   gat_Wo, gat_bo, AGu, B);
  k_attn33<<<t33, 64, 0, stream>>>(QPg, QIF, QF, gat_Wk, gat_bk, gat_Wv, gat_bv,
                                   gat_Wo, gat_bo, AGq, B);
  k_mlp_cat<<<t4B, 128, 0, stream>>>(UF, AGu, out_W1, out_b1, out_W2, out_b2, UE, B);
  k_mlp_cat<<<t4B, 128, 0, stream>>>(QF, AGq, out_W1, out_b1, out_W2, out_b2, QE, B);
  // item graph aggregate + semantic attention
  k_linear64<<<t8B, 256, 0, stream>>>(IFt, UF, gat_Wq, gat_bq, QPig, B);
  k_linear64<<<t8B, 256, 0, stream>>>(IFt, UF, sem_Wq, sem_bq, QPsm, B);
  k_attn33<<<t33, 64, 0, stream>>>(QPig, MUF, IFt, gat_Wk, gat_bk, gat_Wv, gat_bv,
                                   gat_Wo, gat_bo, AU, B);
  k_attn33<<<t33, 64, 0, stream>>>(QPig, MQF, IFt, gat_Wk, gat_bk, gat_Wv, gat_bv,
                                   gat_Wo, gat_bo, AQ, B);
  k_attn2<<<t2B, 64, 0, stream>>>(QPsm, 0, nullptr, 0, nullptr, AU, AQ,
                                  sem_Wk, sem_bk, sem_Wv, sem_bv, sem_Wo, sem_bo, IE, B);
  // head
  k_mlp_cat<<<t4B, 128, 0, stream>>>(UE, QE, qu_W1, qu_b1, qu_W2, qu_b2, QU, B);
  k_dssm<<<t2B, 64, 0, stream>>>(QU, IE, dssm_W1, dssm_b1, dssm_W2, dssm_b2,
                                 (float*)d_out, B);
}
